// BipartiteGraphAutoEncoder_ReportBased_66503273611881
// MI455X (gfx1250) — compile-verified, matrix operand order checked by reading, not run on hardware
//
#include <hip/hip_runtime.h>
#include <math.h>

// ---------------------------------------------------------------------------
// BipartiteGraphAutoEncoder for MI455X (gfx1250, wave32, WMMA)
// Dense GEMMs -> v_wmma_f32_16x16x32_bf16 (f32 accumulate), templated on
// <K, HOUT, ACT, HAS_BIAS> so K-loops fully unroll and epilogues are
// branch-free.
//   - streaming A tile: GLOBAL_LOAD_ASYNC_TO_LDS_B128 (HBM->LDS DMA, ASYNCcnt)
//   - W tile: staged transposed as bf16 (small, L2 resident)
// Graph softmax/scatter -> L2-resident f32 atomics (192MB L2 holds all node
// feature arrays). Edge projection chunked so producer GEMM output is consumed
// from L2 by the scatter kernel.
// ---------------------------------------------------------------------------

#define HD   128     // hidden dim H
#define EDD  32      // edge attr dim
#define LDIM 64      // latent dim
#define ECHUNK 65536 // edges per projection chunk

typedef __attribute__((ext_vector_type(16))) __bf16 v16bf;
typedef __attribute__((ext_vector_type(8)))  float  v8f;

template <int ACT>
__device__ __forceinline__ float apply_act(float x) {
  if (ACT == 1) return x > 0.f ? x : (expf(x) - 1.f); // ELU
  if (ACT == 2) return x > 0.f ? x : 0.f;             // ReLU
  return x;
}

__device__ __forceinline__ void wait_asynccnt0() {
#if __has_builtin(__builtin_amdgcn_s_wait_asynccnt)
  __builtin_amdgcn_s_wait_asynccnt(0);
#else
  asm volatile("s_wait_asynccnt 0x0" ::: "memory");
#endif
}

// ---------------------------------------------------------------------------
// WMMA GEMM:  Out[N x HOUT] = act(A[N x K] @ W[K x HOUT] + bias)
// blockDim.x == HOUT (HOUT/32 waves); each block does 64 rows; each wave owns
// two 16-col tiles. A staged raw f32 via async DMA; W^T staged bf16.
// ---------------------------------------------------------------------------
template <int K, int HOUT, int ACT, bool HAS_BIAS>
__global__ __launch_bounds__(128) void gemm_bias_act(
    const float* __restrict__ A, const float* __restrict__ W,
    const float* __restrict__ bias, float* __restrict__ Out, int N) {
  __shared__ float  Asf[64 * K];    // raw f32 A tile
  __shared__ __bf16 Bt[HOUT * K];   // W transposed: [n][k] bf16
  const int rbase = blockIdx.x * 64;

  // ---- stage A tile: async DMA global->LDS for full tiles ----
  if (rbase + 64 <= N) {
    const unsigned lds_base = (unsigned)(unsigned long long)(&Asf[0]);
    const char* gbase = (const char*)A + (size_t)rbase * K * 4;
#pragma unroll
    for (int b = 0; b < 64 * K * 4; b += HOUT * 16) {
      const int byteOff = b + threadIdx.x * 16;
      unsigned lds = lds_base + (unsigned)byteOff;
      const char* g = gbase + byteOff;
      asm volatile("global_load_async_to_lds_b128 %0, %1, off"
                   :: "v"(lds), "v"(g) : "memory");
    }
  } else {
    for (int idx = threadIdx.x; idx < 64 * K; idx += HOUT) {
      int r = idx / K;
      int grow = rbase + r;
      int k = idx - r * K;
      Asf[idx] = (grow < N) ? A[(size_t)grow * K + k] : 0.f;
    }
  }
  // ---- stage W transposed as bf16 (needs transpose -> VALU path) ----
#pragma unroll 4
  for (int idx = threadIdx.x; idx < HOUT * K; idx += HOUT) {
    int n = idx / K, k = idx - n * K;
    Bt[n * K + k] = (__bf16)W[(size_t)k * HOUT + n];
  }
  wait_asynccnt0();
  __syncthreads();

  const int lane = threadIdx.x & 31;
  const int wave = threadIdx.x >> 5;
  const int nloc = lane & 15;
  const int hi   = lane >> 4;
  const int n0 = (wave * 2) * 16 + nloc;
  const int n1 = (wave * 2 + 1) * 16 + nloc;
  const int c0 = wave * 32 + nloc;
  const int c1 = c0 + 16;
  const float bv0 = HAS_BIAS ? bias[c0] : 0.f;
  const float bv1 = HAS_BIAS ? bias[c1] : 0.f;

  for (int rt = 0; rt < 4; ++rt) {
    v8f acc0 = {0.f,0.f,0.f,0.f,0.f,0.f,0.f,0.f};
    v8f acc1 = {0.f,0.f,0.f,0.f,0.f,0.f,0.f,0.f};
    const int m = rt * 16 + nloc;
#pragma unroll
    for (int kt = 0; kt < K / 32; ++kt) {
      v16bf af, b0, b1;
#pragma unroll
      for (int v = 0; v < 8; ++v) {
        const int k = (kt << 5) + ((v >= 4) ? 16 : 0) + hi * 8 + (v & 3) * 2;
        af[2*v]   = (__bf16)Asf[m * K + k];      // cvt co-executes with WMMA
        af[2*v+1] = (__bf16)Asf[m * K + k + 1];
        b0[2*v]   = Bt[n0 * K + k];
        b0[2*v+1] = Bt[n0 * K + k + 1];
        b1[2*v]   = Bt[n1 * K + k];
        b1[2*v+1] = Bt[n1 * K + k + 1];
      }
      acc0 = __builtin_amdgcn_wmma_f32_16x16x32_bf16(false, af, false, b0,
                                                     (short)0, acc0, false, false);
      acc1 = __builtin_amdgcn_wmma_f32_16x16x32_bf16(false, af, false, b1,
                                                     (short)0, acc1, false, false);
    }
#pragma unroll
    for (int r = 0; r < 8; ++r) {
      const int grow = rbase + rt * 16 + hi * 8 + r;
      if (grow < N) {
        Out[(size_t)grow * HOUT + c0] = apply_act<ACT>(acc0[r] + bv0);
        Out[(size_t)grow * HOUT + c1] = apply_act<ACT>(acc1[r] + bv1);
      }
    }
  }
}

template <int K, int HOUT, int ACT, bool HAS_BIAS>
static inline void gemm_t(const float* A, const float* W, const float* bias,
                          float* Out, int N, hipStream_t stream) {
  gemm_bias_act<K, HOUT, ACT, HAS_BIAS>
      <<<dim3((N + 63) / 64), dim3(HOUT), 0, stream>>>(A, W, bias, Out, N);
}

// fold[k] = sum_h W[k,h] * att[h]   (W row-major K x Hdim)
__global__ void fold_vec_kernel(const float* __restrict__ W,
                                const float* __restrict__ att,
                                float* __restrict__ out, int K, int Hdim) {
  int k = blockIdx.x * blockDim.x + threadIdx.x;
  if (k < K) {
    float acc = 0.f;
    for (int h = 0; h < Hdim; ++h) acc += W[(size_t)k * Hdim + h] * att[h];
    out[k] = acc;
  }
}

// s[n] = dot(X[n,:], vec)  (one wave32 per row)
__global__ void node_dot_kernel(const float* __restrict__ X,
                                const float* __restrict__ vec,
                                float* __restrict__ s, int N, int D) {
  int gw   = (int)((blockIdx.x * (size_t)blockDim.x + threadIdx.x) >> 5);
  int lane = threadIdx.x & 31;
  if (gw >= N) return;
  float acc = 0.f;
  for (int d = lane; d < D; d += 32) acc += X[(size_t)gw * D + d] * vec[d];
  for (int off = 16; off > 0; off >>= 1) acc += __shfl_down(acc, off, 32);
  if (lane == 0) s[gw] = acc;
}

__global__ void init_softmax_kernel(float* __restrict__ m,
                                    float* __restrict__ denom, int n) {
  int i = blockIdx.x * blockDim.x + threadIdx.x;
  if (i < n) { m[i] = -3.402823466e38f; denom[i] = 0.f; }
}

__global__ void zero_kernel(float* __restrict__ p, size_t n) {
  size_t i = (size_t)blockIdx.x * blockDim.x + threadIdx.x;
  size_t stride = (size_t)gridDim.x * blockDim.x;
  for (; i < n; i += stride) p[i] = 0.f;
}

// float atomic max via ordered-int trick (init to -inf)
__device__ __forceinline__ void atomicMaxFloat(float* addr, float val) {
  if (val >= 0.f) atomicMax((int*)addr, __float_as_int(val));
  else            atomicMin((unsigned int*)addr, __float_as_uint(val));
}

__global__ void edge_alpha_max_kernel(const int* __restrict__ src,
                                      const int* __restrict__ dst,
                                      const float* __restrict__ s_src,
                                      const float* __restrict__ s_dst,
                                      const float* __restrict__ s_e,
                                      float* __restrict__ alpha,
                                      float* __restrict__ m, int E) {
  int e = blockIdx.x * blockDim.x + threadIdx.x;
  if (e >= E) return;
  float al = s_src[src[e]] + s_dst[dst[e]] + s_e[e];
  al = (al > 0.f) ? al : 0.01f * al;   // leaky_relu(0.01)
  alpha[e] = al;
  atomicMaxFloat(&m[dst[e]], al);
}

__global__ void edge_exp_sum_kernel(const int* __restrict__ dst,
                                    const float* __restrict__ alpha,
                                    const float* __restrict__ m,
                                    float* __restrict__ a,
                                    float* __restrict__ denom, int E) {
  int e = blockIdx.x * blockDim.x + threadIdx.x;
  if (e >= E) return;
  float v = expf(alpha[e] - m[dst[e]]);
  a[e] = v;
  atomicAdd(&denom[dst[e]], v);
}

// one wave32 per edge: out[dst] += a/denom * (xs[src] + ep[e])   (128 floats)
__global__ void scatter_msg_kernel(const int* __restrict__ src,
                                   const int* __restrict__ dst,
                                   const float* __restrict__ a,
                                   const float* __restrict__ denom,
                                   const float* __restrict__ xs,
                                   const float* __restrict__ ep,
                                   float* __restrict__ out, int e0, int ecnt) {
  int w    = (int)((blockIdx.x * (size_t)blockDim.x + threadIdx.x) >> 5);
  int lane = threadIdx.x & 31;
  if (w >= ecnt) return;
  int e = e0 + w;
  int d = dst[e], s = src[e];
  float c = a[e] / (denom[d] + 1e-16f);
  const float* xrow = xs + (size_t)s * HD;
  const float* erow = ep + (size_t)w * HD;
  float*       orow = out + (size_t)d * HD;
  __builtin_prefetch(xrow, 0, 0);
#pragma unroll
  for (int i = 0; i < 4; ++i) {
    int h = lane + i * 32;
    atomicAdd(&orow[h], c * (xrow[h] + erow[h]));
  }
}

__global__ void residual_elu_kernel(float* __restrict__ acc,
                                    const float* __restrict__ old, size_t n) {
  size_t i = (size_t)blockIdx.x * blockDim.x + threadIdx.x;
  size_t stride = (size_t)gridDim.x * blockDim.x;
  for (; i < n; i += stride) {
    float v = acc[i] + old[i];
    acc[i] = v > 0.f ? v : (expf(v) - 1.f);
  }
}

// ---------------------------------------------------------------------------
extern "C" void kernel_launch(void* const* d_in, const int* in_sizes, int n_in,
                              void* d_out, int out_size, void* d_ws, size_t ws_size,
                              hipStream_t stream) {
  const float* x_member   = (const float*)d_in[0];
  const float* x_provider = (const float*)d_in[1];
  const int*   edge_index = (const int*)d_in[2];
  const float* edge_attr  = (const float*)d_in[3];
  const float* W_proj_m = (const float*)d_in[4];
  const float* b_proj_m = (const float*)d_in[5];
  const float* W_proj_p = (const float*)d_in[6];
  const float* b_proj_p = (const float*)d_in[7];
  const float* W_conv   = (const float*)d_in[8];   // [2,2,2,H,H]
  const float* W_conv_e = (const float*)d_in[9];   // [2,2,ED,H]
  const float* att_conv = (const float*)d_in[10];  // [2,2,3,H]
  const float* W_fin_m  = (const float*)d_in[11];
  const float* b_fin_m  = (const float*)d_in[12];
  const float* W_fin_p  = (const float*)d_in[13];
  const float* b_fin_p  = (const float*)d_in[14];
  const float* Wd_m1 = (const float*)d_in[15];
  const float* bd_m1 = (const float*)d_in[16];
  const float* Wd_m2 = (const float*)d_in[17];
  const float* bd_m2 = (const float*)d_in[18];
  const float* Wd_p1 = (const float*)d_in[19];
  const float* bd_p1 = (const float*)d_in[20];
  const float* Wd_p2 = (const float*)d_in[21];
  const float* bd_p2 = (const float*)d_in[22];

  const int N_M = in_sizes[0] / HD;
  const int N_P = in_sizes[1] / HD;
  const int E   = in_sizes[2] / 2;
  const int* src = edge_index;       // provider ids
  const int* dst = edge_index + E;   // member ids

  // ---- workspace layout (floats) ----
  float* ws = (float*)d_ws;
  size_t off = 0;
  float* hm0    = ws + off; off += (size_t)N_M * HD;
  float* hm1    = ws + off; off += (size_t)N_M * HD;
  float* hp0    = ws + off; off += (size_t)N_P * HD;
  float* hp1    = ws + off; off += (size_t)N_P * HD;
  float* xsbuf  = ws + off; off += (size_t)N_M * HD;   // src projections / decoder hidden
  float* epchk  = ws + off; off += (size_t)ECHUNK * HD;
  float* alpha  = ws + off; off += (size_t)E;
  float* aexp   = ws + off; off += (size_t)E;
  float* s_e    = ws + off; off += (size_t)E;
  float* mmax   = ws + off; off += (size_t)N_M;
  float* denom  = ws + off; off += (size_t)N_M;
  float* s_src  = ws + off; off += (size_t)N_M;
  float* s_dst  = ws + off; off += (size_t)N_M;
  float* fold_s = ws + off; off += HD;
  float* fold_d = ws + off; off += HD;
  float* fold_e = ws + off; off += HD;
  (void)ws_size; (void)n_in;

  // ---- input projections ----
  gemm_t<128, 128, 1, true>(x_member,   W_proj_m, b_proj_m, hm0, N_M, stream);
  gemm_t<128, 128, 1, true>(x_provider, W_proj_p, b_proj_p, hp0, N_P, stream);

  float *hmA = hm0, *hmB = hm1, *hpA = hp0, *hpB = hp1;
  for (int l = 0; l < 2; ++l) {
    for (int dir = 0; dir < 2; ++dir) {
      const float *xsrc, *xdst; const int *sarr, *darr;
      int Ns, Nd; float* accum;
      if (dir == 0) { xsrc = hpA; Ns = N_P; xdst = hmA; Nd = N_M;
                      sarr = src; darr = dst; accum = hmB; }
      else          { xsrc = hmA; Ns = N_M; xdst = hpA; Nd = N_P;
                      sarr = dst; darr = src; accum = hpB; }
      const float* Wsd  = W_conv + (size_t)((l * 2 + dir) * 2) * HD * HD;
      const float* Wsrc = Wsd;
      const float* Wdst = Wsd + HD * HD;
      const float* We   = W_conv_e + (size_t)(l * 2 + dir) * EDD * HD;
      const float* att  = att_conv + (size_t)(l * 2 + dir) * 3 * HD;

      // fold attention vectors through weights: s = x . (W @ att)
      fold_vec_kernel<<<1, 128, 0, stream>>>(Wsrc, att,        fold_s, HD,  HD);
      fold_vec_kernel<<<1, 128, 0, stream>>>(Wdst, att + HD,   fold_d, HD,  HD);
      fold_vec_kernel<<<1, 128, 0, stream>>>(We,   att + 2*HD, fold_e, EDD, HD);

      node_dot_kernel<<<(Ns + 7) / 8, 256, 0, stream>>>(xsrc,      fold_s, s_src, Ns, HD);
      node_dot_kernel<<<(Nd + 7) / 8, 256, 0, stream>>>(xdst,      fold_d, s_dst, Nd, HD);
      node_dot_kernel<<<(E  + 7) / 8, 256, 0, stream>>>(edge_attr, fold_e, s_e,   E,  EDD);

      // segment softmax over dst
      init_softmax_kernel<<<(Nd + 255) / 256, 256, 0, stream>>>(mmax, denom, Nd);
      edge_alpha_max_kernel<<<(E + 255) / 256, 256, 0, stream>>>(
          sarr, darr, s_src, s_dst, s_e, alpha, mmax, E);
      edge_exp_sum_kernel<<<(E + 255) / 256, 256, 0, stream>>>(
          darr, alpha, mmax, aexp, denom, E);

      // source projection (WMMA) + chunked edge projection (WMMA) + scatter
      gemm_t<128, 128, 0, false>(xsrc, Wsrc, nullptr, xsbuf, Ns, stream);
      zero_kernel<<<2048, 256, 0, stream>>>(accum, (size_t)Nd * HD);
      for (int e0 = 0; e0 < E; e0 += ECHUNK) {
        int ecnt = (E - e0 < ECHUNK) ? (E - e0) : ECHUNK;
        gemm_t<32, 128, 0, false>(edge_attr + (size_t)e0 * EDD, We, nullptr,
                                  epchk, ecnt, stream);
        scatter_msg_kernel<<<(ecnt + 7) / 8, 256, 0, stream>>>(
            sarr, darr, aexp, denom, xsbuf, epchk, accum, e0, ecnt);
      }
    }
    residual_elu_kernel<<<2048, 256, 0, stream>>>(hmB, hmA, (size_t)N_M * HD);
    residual_elu_kernel<<<2048, 256, 0, stream>>>(hpB, hpA, (size_t)N_P * HD);
    float* t;
    t = hmA; hmA = hmB; hmB = t;
    t = hpA; hpA = hpB; hpB = t;
  }

  // ---- heads + decoders (outputs: x_hat_m, x_hat_p, z_m, z_p concatenated) ----
  float* out_xm = (float*)d_out;
  float* out_xp = out_xm + (size_t)N_M * HD;
  float* out_zm = out_xp + (size_t)N_P * HD;
  float* out_zp = out_zm + (size_t)N_M * LDIM;

  gemm_t<128, 64, 0, true>(hmA, W_fin_m, b_fin_m, out_zm, N_M, stream);
  gemm_t<128, 64, 0, true>(hpA, W_fin_p, b_fin_p, out_zp, N_P, stream);

  gemm_t<64, 128, 2, true>(out_zm, Wd_m1, bd_m1, xsbuf, N_M, stream);   // relu
  gemm_t<128, 128, 0, true>(xsbuf, Wd_m2, bd_m2, out_xm, N_M, stream);
  gemm_t<64, 128, 2, true>(out_zp, Wd_p1, bd_p1, xsbuf, N_P, stream);   // relu
  gemm_t<128, 128, 0, true>(xsbuf, Wd_p2, bd_p2, out_xp, N_P, stream);
}
